// QuantizedConv2d8bit_74594991997152
// MI455X (gfx1250) — matrix operand order repaired and checked by reading, not compiled, not run on hardware
//
#include <hip/hip_runtime.h>
#include <hip/hip_bf16.h>
#include <stddef.h>

// ---------------------------------------------------------------------------
// Quantized 3x3 conv (B=32, Cin=Cout=256, 56x56, pad 1) as implicit GEMM on
// CDNA5 WMMA. M = 100352 pixels, N = 256, K = 2304.
//   FLOPs ~ 118.4 G, min HBM bytes ~ 210 MB -> ~565 FLOP/B -> compute bound.
// v_wmma_f32_16x16x32_f16: int8 weights (q - zp) exact in f16; per-channel
// scale + bias in f32 epilogue; only loss is x -> f16 rounding.
// Each wave computes a 32M x 64N tile: 2 A fragments amortize every B
// fragment (8 WMMA : 12 b128 loads per k-step).
// ---------------------------------------------------------------------------

typedef __attribute__((ext_vector_type(16))) _Float16 v16h;
typedef __attribute__((ext_vector_type(8)))  _Float16 v8h;
typedef __attribute__((ext_vector_type(8)))  float    v8f;

#define Bn    32
#define CIN   256
#define COUT  256
#define Hh    56
#define Ww    56
#define HP    58              // padded
#define WP    58
#define HWo   (Hh * Ww)       // 3136 (divisible by 32)
#define KTOT  (CIN * 9)       // 2304
#define NPIX_PAD (Bn * HP * WP)           // 107648 padded pixels
#define XP_ELEMS ((size_t)NPIX_PAD * CIN) // f16 elements
#define XP_BYTES (XP_ELEMS * 2)           // 55,115,776 (512B aligned)

// ---------------------------------------------------------------------------
// Kernel 1: NCHW f32 -> zero-padded NHWC f16. One thread per padded pixel.
// ---------------------------------------------------------------------------
__global__ void qc_pack_x(const float* __restrict__ x, _Float16* __restrict__ Xp) {
    int idx = blockIdx.x * 256 + threadIdx.x;
    if (idx >= NPIX_PAD) return;
    int b   = idx / (HP * WP);
    int rem = idx - b * (HP * WP);
    int hp  = rem / WP;
    int wp  = rem - hp * WP;
    _Float16* dst = Xp + (size_t)idx * CIN;
    if (hp >= 1 && hp <= Hh && wp >= 1 && wp <= Ww) {
        const float* src = x + (size_t)b * CIN * HWo + (hp - 1) * Ww + (wp - 1);
        #pragma unroll 4
        for (int ci = 0; ci < CIN; ci += 8) {
            v8h v;
            #pragma unroll
            for (int j = 0; j < 8; ++j)
                v[j] = (_Float16)src[(size_t)(ci + j) * HWo];
            *(v8h*)(dst + ci) = v;
        }
    } else {
        v8h z = {};
        #pragma unroll 4
        for (int ci = 0; ci < CIN; ci += 8)
            *(v8h*)(dst + ci) = z;
    }
}

// ---------------------------------------------------------------------------
// Kernel 2: OIHW int32 weights -> Wt[co][k] f16, k = (kh*3+kw)*256 + ci.
// Stores (q - zp) exactly in f16; scale applied later in f32.
// ---------------------------------------------------------------------------
__global__ void qc_pack_w(const int* __restrict__ wq, const int* __restrict__ zp,
                          _Float16* __restrict__ Wt) {
    int i = blockIdx.x * 256 + threadIdx.x;
    if (i >= COUT * KTOT) return;
    int co = i / KTOT;
    int k  = i - co * KTOT;
    int kk = k >> 8;          // tap index kh*3+kw
    int ci = k & 255;
    int val = wq[((size_t)co * CIN + ci) * 9 + kk] - zp[co];
    Wt[i] = (_Float16)(float)val;
}

// ---------------------------------------------------------------------------
// Kernel 3: implicit-GEMM conv. 4 waves/block; wave w owns M-tile of 32
// pixels (2 x 16) x 64 cout. 72 k-steps x 8 WMMAs = 576 v_wmma per wave.
// All loads contiguous & aligned (A and B share k = 16*hl + s order); EXEC all-ones.
// ---------------------------------------------------------------------------
__global__ __launch_bounds__(128) void qc_conv_wmma(
    const _Float16* __restrict__ Xp, const _Float16* __restrict__ Wt,
    const float* __restrict__ wscale, const int* __restrict__ biasq,
    const float* __restrict__ bscale, float* __restrict__ out)
{
    const int lane = threadIdx.x & 31;
    const int wid  = threadIdx.x >> 5;
    const int ln   = lane & 15;       // N index / M index within 16-tile
    const int hl   = lane >> 4;       // lane half selects K sub-range

    const int mtile = blockIdx.x * 4 + wid;   // 32-pixel M tile
    const int m0    = mtile * 32;
    const int n0    = blockIdx.y * 64;

    // All 32 pixels share b (3136 % 32 == 0)
    const int b    = m0 / HWo;
    const int rem0 = m0 - b * HWo;

    // Per-lane A pixels for the two 16-row halves
    const int p0 = rem0 + ln;
    const int h0 = p0 / Ww, w0 = p0 - h0 * Ww;
    const int p1 = rem0 + 16 + ln;
    const int h1 = p1 / Ww, w1 = p1 - h1 * Ww;

    const _Float16* xbase0 =
        Xp + ((size_t)(b * HP + h0) * WP + w0) * CIN + 16 * hl;
    const _Float16* xbase1 =
        Xp + ((size_t)(b * HP + h1) * WP + w1) * CIN + 16 * hl;

    // B rows: lane ln handles co = n0 + nb*16 + ln; K-run starts at 16*hl
    const _Float16* wrow[4];
    #pragma unroll
    for (int nb = 0; nb < 4; ++nb)
        wrow[nb] = Wt + (size_t)(n0 + nb * 16 + ln) * KTOT + 16 * hl;

    v8f accA[4] = {};   // M rows 0..15 of the tile
    v8f accB[4] = {};   // M rows 16..31

    for (int kk = 0; kk < 9; ++kk) {           // 3x3 taps
        const int kh = kk / 3;
        const int kw = kk - kh * 3;
        const size_t toff = (size_t)(kh * WP + kw) * CIN;
        const _Float16* xt0 = xbase0 + toff;
        const _Float16* xt1 = xbase1 + toff;
        const int kb0 = kk * CIN;
        #pragma unroll 2
        for (int kc = 0; kc < CIN; kc += 32) { // K chunk of 32
            // A fragments: lanes 0-15 K {kc..+15},
            //              lanes 16-31 K {kc+16..+31}
            v8h a0l = *(const v8h*)(xt0 + kc);
            v8h a0h = *(const v8h*)(xt0 + kc + 8);
            v8h a1l = *(const v8h*)(xt1 + kc);
            v8h a1h = *(const v8h*)(xt1 + kc + 8);
            v16h a0 = __builtin_shufflevector(a0l, a0h,
                      0,1,2,3,4,5,6,7, 8,9,10,11,12,13,14,15);
            v16h a1 = __builtin_shufflevector(a1l, a1h,
                      0,1,2,3,4,5,6,7, 8,9,10,11,12,13,14,15);
            #pragma unroll
            for (int nb = 0; nb < 4; ++nb) {
                v16h bf = *(const v16h*)(wrow[nb] + kb0 + kc);
                accA[nb] = __builtin_amdgcn_wmma_f32_16x16x32_f16(
                    false, a0, false, bf, (short)0, accA[nb], false, false);
                accB[nb] = __builtin_amdgcn_wmma_f32_16x16x32_f16(
                    false, a1, false, bf, (short)0, accB[nb], false, false);
            }
        }
    }

    // Epilogue: per-channel dequant scale + bias, aligned 32B stores.
    // C/D layout: lane ln = N, vgpr r -> M = r + 8*hl.
    const float bsc = bscale[0];
    #pragma unroll
    for (int nb = 0; nb < 4; ++nb) {
        const int co = n0 + nb * 16 + ln;
        const float s  = wscale[co];
        const float bv = bsc * (float)biasq[co];
        float* obase = out + (size_t)(b * COUT + co) * HWo + rem0 + 8 * hl;
        v8f rA, rB;
        #pragma unroll
        for (int j = 0; j < 8; ++j) {
            rA[j] = accA[nb][j] * s + bv;
            rB[j] = accB[nb][j] * s + bv;
        }
        *(v8f*)(obase)      = rA;   // rows 0..15 half
        *(v8f*)(obase + 16) = rB;   // rows 16..31 half
    }
}

// ---------------------------------------------------------------------------
extern "C" void kernel_launch(void* const* d_in, const int* in_sizes, int n_in,
                              void* d_out, int out_size, void* d_ws, size_t ws_size,
                              hipStream_t stream) {
    const float* x      = (const float*)d_in[0];
    const int*   wq     = (const int*)  d_in[1];
    const float* wscale = (const float*)d_in[2];
    const int*   zp     = (const int*)  d_in[3];
    const int*   biasq  = (const int*)  d_in[4];
    const float* bscale = (const float*)d_in[5];
    float* out = (float*)d_out;

    _Float16* Xp = (_Float16*)d_ws;                          // 55.1 MB
    _Float16* Wt = (_Float16*)((char*)d_ws + XP_BYTES);      // +1.18 MB

    (void)in_sizes; (void)n_in; (void)out_size; (void)ws_size;

    // 1) pad + convert activations
    qc_pack_x<<<(NPIX_PAD + 255) / 256, 256, 0, stream>>>(x, Xp);
    // 2) repack weights (exact int in f16)
    qc_pack_w<<<(COUT * KTOT) / 256, 256, 0, stream>>>(wq, zp, Wt);
    // 3) implicit-GEMM conv: 3136 M-tiles(32) / 4 waves per block, 4 N-groups
    dim3 grid((Bn * HWo / 32) / 4, COUT / 64);               // (784, 4)
    qc_conv_wmma<<<grid, 128, 0, stream>>>(Xp, Wt, wscale, biasq, bscale, out);
}
